// MultiHeadSelfAttention_21096879358375
// MI455X (gfx1250) — matrix operand order repaired and checked by reading, not compiled, not run on hardware
//
#include <hip/hip_runtime.h>
#include <hip/hip_bf16.h>

// MI455X / gfx1250 multi-head self-attention, bf16 WMMA + TDM tile staging.
// Workspace (bf16), ~92.3 MB: xb[8192,1024], wq/wk/wv/wo_b[1024,1024],
// qb,kb [B,S,D], vtb [B,H,Dh,S], ctxb [B,S,D].

typedef __bf16 bf16;
typedef __attribute__((ext_vector_type(16))) __bf16 bf16x16;
typedef __attribute__((ext_vector_type(8)))  __bf16 bf16x8;
typedef __attribute__((ext_vector_type(8)))  float  f32x8;
typedef __attribute__((ext_vector_type(4)))  unsigned int u32x4;
typedef __attribute__((ext_vector_type(8)))  int i32x8;
typedef __attribute__((ext_vector_type(4)))  int i32x4;

#define NEG_INF (-3.402823466e38f)

#if defined(__has_builtin)
# if __has_builtin(__builtin_amdgcn_tensor_load_to_lds) && \
     __has_builtin(__builtin_amdgcn_s_wait_tensorcnt)
#  define HAVE_TDM 1
# endif
# if __has_builtin(__builtin_amdgcn_permlane16)
#  define HAVE_PERMLANE 1
# endif
#endif
#ifndef HAVE_TDM
# define HAVE_TDM 0
#endif
#ifndef HAVE_PERMLANE
# define HAVE_PERMLANE 0
#endif

__device__ __forceinline__ f32x8 wmma_bf16(bf16x16 a, bf16x16 b, f32x8 c) {
  return __builtin_amdgcn_wmma_f32_16x16x32_bf16(
      false, a, false, b, (short)0, c, false, false);
}

// A-matrix 16x32 (MxK) fragment, wave32 layout:
//   lane l: m = l&15, kb = (l>>4)*8; elems 0-7 -> k=kb.., elems 8-15 -> k=16+kb..
__device__ __forceinline__ bf16x16 load_a_frag(const bf16* base, int row_stride, int lane) {
  const bf16* p = base + (size_t)(lane & 15) * row_stride + ((lane >> 4) * 8);
  union { bf16x16 v; bf16x8 h[2]; } u;
  u.h[0] = *(const bf16x8*)(p);
  u.h[1] = *(const bf16x8*)(p + 16);
  return u.v;
}

// B-matrix 32x16 (KxN) fragment from a [N,K]-major source:
//   lane l: n = l&15 (source row), kb = (l>>4)*8; elems 0-7 -> k=kb.., elems 8-15 -> k=16+kb..
__device__ __forceinline__ bf16x16 load_b_frag(const bf16* base, int row_stride, int lane) {
  const bf16* p = base + (size_t)(lane & 15) * row_stride + ((lane >> 4) * 8);
  union { bf16x16 v; bf16x8 h[2]; } u;
  u.h[0] = *(const bf16x8*)(p);
  u.h[1] = *(const bf16x8*)(p + 16);
  return u.v;
}

// max with lane (lane^m) inside each 16-lane half: v_permlane16_b32 butterfly
__device__ __forceinline__ float xor16_max(float v, int mask, unsigned s0, unsigned s1) {
#if HAVE_PERMLANE
  union { float f; unsigned u; } a, b;
  a.f = v;
  b.u = __builtin_amdgcn_permlane16(a.u, a.u, s0, s1, false, false);
  return fmaxf(v, b.f);
#else
  return fmaxf(v, __shfl_xor(v, mask));
#endif
}

#if HAVE_TDM
// TENSOR_LOAD_TO_LDS of a 2-D bf16 tile [tile_d1 rows x tile_d0 elems]
// from a row-major tensor (row stride stride0 elems) into LDS at lds_off.
__device__ __forceinline__ void tdm_load_tile(unsigned lds_off, const bf16* gaddr,
                                              unsigned tensor_d0, unsigned tensor_d1,
                                              unsigned tile_d0, unsigned tile_d1,
                                              unsigned stride0) {
  unsigned long long ga = (unsigned long long)(uintptr_t)gaddr;
  u32x4 g0 = {};
  g0[0] = 1u;                                          // count=1, user D#
  g0[1] = lds_off;                                     // lds_addr (bytes)
  g0[2] = (unsigned)(ga & 0xffffffffu);                // global_addr[31:0]
  g0[3] = (unsigned)((ga >> 32) & 0x01ffffffu) | (2u << 30);  // addr[56:32] | type=2
  i32x8 g1 = {};
  g1[0] = 1 << 16;                                     // data_size = 2 bytes
  g1[1] = (int)((tensor_d0 & 0xffffu) << 16);          // tensor_dim0[15:0]
  g1[2] = (int)((tensor_d0 >> 16) | ((tensor_d1 & 0xffffu) << 16));
  g1[3] = (int)((tensor_d1 >> 16) | ((tile_d0 & 0xffffu) << 16));
  g1[4] = (int)(tile_d1 & 0xffffu);                    // tile_dim1 (tile_dim2=0)
  g1[5] = (int)stride0;                                // tensor_dim0_stride[31:0]
  g1[6] = 0; g1[7] = 0;
  i32x4 z = {};
#if __clang_major__ >= 23
  i32x8 z8 = {};
  __builtin_amdgcn_tensor_load_to_lds(g0, g1, z, z, z8, 0);
#else
  __builtin_amdgcn_tensor_load_to_lds(g0, g1, z, z, 0);
#endif
}
#endif

// ---------------------------------------------------------------------------
// fp32 -> bf16 conversion, 8 elements per thread
// ---------------------------------------------------------------------------
__global__ void cvt_f32_bf16(const float* __restrict__ src, bf16* __restrict__ dst, int n) {
  int i = (blockIdx.x * blockDim.x + threadIdx.x) * 8;
  if (i < n) {
    float4 f0 = *(const float4*)(src + i);
    float4 f1 = *(const float4*)(src + i + 4);
    union { bf16x8 v; bf16 e[8]; } u;
    u.e[0] = (bf16)f0.x; u.e[1] = (bf16)f0.y; u.e[2] = (bf16)f0.z; u.e[3] = (bf16)f0.w;
    u.e[4] = (bf16)f1.x; u.e[5] = (bf16)f1.y; u.e[6] = (bf16)f1.z; u.e[7] = (bf16)f1.w;
    *(bf16x8*)(dst + i) = u.v;
  }
}

// ---------------------------------------------------------------------------
// GEMM: OUT = A[M,K] * W[N,K]^T + bias, M=8192, N=K=1024.
// Block tile 128x128 (8 waves 4x2), wave tile 32x64.
// TDM path: 128x64 bf16 A/B slabs (16KB each) double-buffered in LDS via
// TENSOR_LOAD_TO_LDS (wave 0), s_wait_tensorcnt + barrier; 16 WMMAs per sync.
// OUT_MODE 0: bf16 [M,N]   1: bf16 V-transposed [B,H,Dh,S]   2: f32 [M,N]
// ---------------------------------------------------------------------------
template <int OUT_MODE>
__global__ __launch_bounds__(256)
void gemm_bf16_wmma(const bf16* __restrict__ A, const bf16* __restrict__ W,
                    const float* __restrict__ bias, void* __restrict__ out) {
  constexpr int N = 1024, K = 1024;
  const int lane = threadIdx.x & 31;
  const int wave = threadIdx.x >> 5;
  const int m0l = (wave & 3) * 32;                 // local row within 128-row block
  const int n0l = (wave >> 2) * 64;                // local col within 128-col block
  const int m0g = blockIdx.y * 128;
  const int n0g = blockIdx.x * 128;

  f32x8 acc[2][4] = {};

#if HAVE_TDM
  constexpr int KSTEP = 64;
  constexpr unsigned TILE_B = 128 * KSTEP * 2;     // 16KB per slab
  __shared__ __align__(16) bf16 atile[2][128 * KSTEP];
  __shared__ __align__(16) bf16 btile[2][128 * KSTEP];
  const unsigned a_lds = (unsigned)(uintptr_t)&atile[0][0];
  const unsigned b_lds = (unsigned)(uintptr_t)&btile[0][0];

  if (wave == 0) {
    tdm_load_tile(a_lds, A + (size_t)m0g * K, K, 8192, KSTEP, 128, K);
    tdm_load_tile(b_lds, W + (size_t)n0g * K, K, 1024, KSTEP, 128, K);
  }
  int buf = 0;
  for (int k = 0; k < K; k += KSTEP) {
    if (wave == 0) {
      if (k + KSTEP < K) {
        tdm_load_tile(a_lds + (buf ^ 1) * TILE_B, A + (size_t)m0g * K + k + KSTEP,
                      K, 8192, KSTEP, 128, K);
        tdm_load_tile(b_lds + (buf ^ 1) * TILE_B, W + (size_t)n0g * K + k + KSTEP,
                      K, 1024, KSTEP, 128, K);
        __builtin_amdgcn_s_wait_tensorcnt((short)2);  // prior pair complete
      } else {
        __builtin_amdgcn_s_wait_tensorcnt((short)0);
      }
    }
    __syncthreads();                               // slabs of this step visible
    const bf16* at = &atile[buf][0];
    const bf16* bt = &btile[buf][0];
#pragma unroll
    for (int ks = 0; ks < KSTEP; ks += 32) {
      bf16x16 a0 = load_a_frag(at + (size_t)m0l * KSTEP + ks, KSTEP, lane);
      bf16x16 a1 = load_a_frag(at + (size_t)(m0l + 16) * KSTEP + ks, KSTEP, lane);
#pragma unroll
      for (int j = 0; j < 4; ++j) {
        bf16x16 b = load_b_frag(bt + (size_t)(n0l + 16 * j) * KSTEP + ks, KSTEP, lane);
        acc[0][j] = wmma_bf16(a0, b, acc[0][j]);
        acc[1][j] = wmma_bf16(a1, b, acc[1][j]);
      }
    }
    __syncthreads();                               // done reading before overwrite
    buf ^= 1;
  }
#else
  for (int k = 0; k < K; k += 32) {
    bf16x16 a0 = load_a_frag(A + (size_t)(m0g + m0l) * K + k, K, lane);
    bf16x16 a1 = load_a_frag(A + (size_t)(m0g + m0l + 16) * K + k, K, lane);
#pragma unroll
    for (int j = 0; j < 4; ++j) {
      bf16x16 b = load_b_frag(W + (size_t)(n0g + n0l + 16 * j) * K + k, K, lane);
      acc[0][j] = wmma_bf16(a0, b, acc[0][j]);
      acc[1][j] = wmma_bf16(a1, b, acc[1][j]);
    }
  }
#endif

  const int m0 = m0g + m0l;
  const int n0 = n0g + n0l;
#pragma unroll
  for (int i = 0; i < 2; ++i) {
#pragma unroll
    for (int j = 0; j < 4; ++j) {
      const int n = n0 + 16 * j + (lane & 15);
      const float bv = bias[n];
      const int mbase = m0 + 16 * i + (lane >> 4) * 8;   // 8 consecutive rows per lane
      if constexpr (OUT_MODE == 0) {
        bf16* o = (bf16*)out;
#pragma unroll
        for (int r = 0; r < 8; ++r)
          o[(size_t)(mbase + r) * N + n] = (bf16)(acc[i][j][r] + bv);
      } else if constexpr (OUT_MODE == 1) {
        bf16* o = (bf16*)out;   // Vt[b,h,d,s]; 8 consecutive s => one 16B store
        const int bidx = mbase >> 11, s = mbase & 2047;
        const int hh = n >> 6, d = n & 63;
        union { bf16x8 v; bf16 e[8]; } u;
#pragma unroll
        for (int r = 0; r < 8; ++r) u.e[r] = (bf16)(acc[i][j][r] + bv);
        *(bf16x8*)(o + ((size_t)((bidx * 16 + hh) * 64 + d)) * 2048 + s) = u.v;
      } else {
        float* o = (float*)out;
#pragma unroll
        for (int r = 0; r < 8; ++r)
          o[(size_t)(mbase + r) * N + n] = acc[i][j][r] + bv;
      }
    }
  }
}

// ---------------------------------------------------------------------------
// Flash attention: block = 128 queries of one (b,h), one wave = 16 queries.
// All 8 waves share the key stream, so K (64x64) and Vt (64x64) tiles are
// TDM-staged into LDS double-buffered by wave 0 (8x fewer global loads).
// Row max via v_permlane16 butterfly; row sums via WMMA against all-ones B.
// Scale 1/sqrt(Dh)=0.125 folded into Q fragments (exact exponent shift).
// ---------------------------------------------------------------------------
__global__ __launch_bounds__(256)
void flash_attn_wmma(const bf16* __restrict__ Q, const bf16* __restrict__ Kb,
                     const bf16* __restrict__ Vt, bf16* __restrict__ ctx) {
  constexpr int S = 2048, D = 1024, Dh = 64, H = 16;
  __shared__ __align__(16) bf16 pbuf[8][16 * 64];   // per-wave P tile (16x64)
#if HAVE_TDM
  __shared__ __align__(16) bf16 ktile[2][64 * 64];  // [key][dh], 8KB each
  __shared__ __align__(16) bf16 vtile[2][64 * 64];  // [dh][key], 8KB each
#endif

  const int lane = threadIdx.x & 31;
  const int wave = threadIdx.x >> 5;
  const int bh = blockIdx.x;                        // 0..63
  const int b = bh >> 4, h = bh & 15;
  const int q0 = blockIdx.y * 128 + wave * 16;

  const bf16* qbase = Q + (size_t)(b * S + q0) * D + h * Dh;
  const bf16* kbase = Kb + (size_t)b * S * D + h * Dh;
  const bf16* vbase = Vt + (size_t)(b * H + h) * Dh * S;

  bf16x16 qf0 = load_a_frag(qbase + 0, D, lane);    // dh 0..31
  bf16x16 qf1 = load_a_frag(qbase + 32, D, lane);   // dh 32..63
  {                                                 // fold 0.125 into Q (exact)
    union { bf16x16 v; bf16 e[16]; } u0, u1;
    u0.v = qf0; u1.v = qf1;
#pragma unroll
    for (int e = 0; e < 16; ++e) {
      u0.e[e] = (bf16)((float)u0.e[e] * 0.125f);
      u1.e[e] = (bf16)((float)u1.e[e] * 0.125f);
    }
    qf0 = u0.v; qf1 = u1.v;
  }

  bf16x16 onesf;                                    // all-ones B fragment
  {
    union { bf16x16 v; bf16 e[16]; } ou;
#pragma unroll
    for (int e = 0; e < 16; ++e) ou.e[e] = (bf16)1.0f;
    onesf = ou.v;
  }

  f32x8 o[4] = {};
  f32x8 ssum = {};                                  // running softmax denominators
  float mrow[8];
#pragma unroll
  for (int r = 0; r < 8; ++r) mrow[r] = NEG_INF;

#if HAVE_TDM
  constexpr unsigned KV_B = 64 * 64 * 2;            // 8KB per tile
  const unsigned k_lds = (unsigned)(uintptr_t)&ktile[0][0];
  const unsigned v_lds = (unsigned)(uintptr_t)&vtile[0][0];
  if (wave == 0) {
    tdm_load_tile(k_lds, kbase, 64, 64, 64, 64, D);
    tdm_load_tile(v_lds, vbase, 64, 64, 64, 64, S);
  }
  int buf = 0;
#endif

  for (int k0 = 0; k0 < S; k0 += 64) {
#if HAVE_TDM
    if (wave == 0) {
      if (k0 + 64 < S) {
        tdm_load_tile(k_lds + (buf ^ 1) * KV_B, kbase + (size_t)(k0 + 64) * D,
                      64, 64, 64, 64, D);
        tdm_load_tile(v_lds + (buf ^ 1) * KV_B, vbase + (k0 + 64),
                      64, 64, 64, 64, S);
        __builtin_amdgcn_s_wait_tensorcnt((short)2);
      } else {
        __builtin_amdgcn_s_wait_tensorcnt((short)0);
      }
    }
    __syncthreads();                                // this step's K/V visible
    const bf16* kroot = &ktile[buf][0]; const int kstride = 64;
    const bf16* vroot = &vtile[buf][0]; const int vstride = 64; const int koff = 0;
#else
    if (k0 + 64 < S) {
      __builtin_prefetch(kbase + (size_t)(k0 + 64 + lane) * D, 0, 0);
      __builtin_prefetch(vbase + (size_t)(lane & 31) * S + k0 + 64, 0, 0);
    }
    const bf16* kroot = kbase + (size_t)k0 * D; const int kstride = D;
    const bf16* vroot = vbase; const int vstride = S; const int koff = k0;
#endif

    // ---- scores: four 16-key tiles (Q pre-scaled) ----
    f32x8 sc[4];
#pragma unroll
    for (int t = 0; t < 4; ++t) {
      const bf16* kr = kroot + (size_t)(16 * t) * kstride;
      f32x8 s = {};
      s = wmma_bf16(qf0, load_b_frag(kr + 0, kstride, lane), s);
      s = wmma_bf16(qf1, load_b_frag(kr + 32, kstride, lane), s);
      sc[t] = s;
    }

    // ---- online softmax: max via permlane16 butterfly; sums via WMMA ----
#pragma unroll
    for (int r = 0; r < 8; ++r) {
      float v = fmaxf(fmaxf(sc[0][r], sc[1][r]), fmaxf(sc[2][r], sc[3][r]));
      v = xor16_max(v, 1, 0x67452301u, 0xEFCDAB89u);
      v = xor16_max(v, 2, 0x54761032u, 0xDCFE98BAu);
      v = xor16_max(v, 4, 0x32107654u, 0xBA98FEDCu);
      v = xor16_max(v, 8, 0xFEDCBA98u, 0x76543210u);
      const float mn = fmaxf(mrow[r], v);
      const float corr = __expf(mrow[r] - mn);
      mrow[r] = mn;
#pragma unroll
      for (int t = 0; t < 4; ++t) sc[t][r] = __expf(sc[t][r] - mn);
      ssum[r] *= corr;
      o[0][r] *= corr; o[1][r] *= corr; o[2][r] *= corr; o[3][r] *= corr;
    }

    // ---- restage P (C-layout -> row-major 16x64 in LDS -> A-fragments) ----
    bf16* pw = pbuf[wave];
    {
      const int M0 = (lane >> 4) * 8;
      const int n = lane & 15;
#pragma unroll
      for (int r = 0; r < 8; ++r) {
#pragma unroll
        for (int t = 0; t < 4; ++t)
          pw[(M0 + r) * 64 + 16 * t + n] = (bf16)sc[t][r];
      }
    }
    bf16x16 pf0 = load_a_frag(pw + 0, 64, lane);    // keys k0..k0+31
    bf16x16 pf1 = load_a_frag(pw + 32, 64, lane);   // keys k0+32..k0+63

    // ---- denominator: ssum += P * ones ----
    ssum = wmma_bf16(pf0, onesf, ssum);
    ssum = wmma_bf16(pf1, onesf, ssum);

    // ---- O += P * V (4 dh tiles x 2 key k-steps) ----
#pragma unroll
    for (int j = 0; j < 4; ++j) {
      const bf16* vr = vroot + (size_t)(16 * j) * vstride + koff;
      o[j] = wmma_bf16(pf0, load_b_frag(vr + 0, vstride, lane), o[j]);
      o[j] = wmma_bf16(pf1, load_b_frag(vr + 32, vstride, lane), o[j]);
    }

#if HAVE_TDM
    __syncthreads();                                // reads done before overwrite
    buf ^= 1;
#endif
  }

  // ---- normalize and write ctx[B,S,D] ----
#pragma unroll
  for (int r = 0; r < 8; ++r) {
    const float inv = 1.0f / ssum[r];
    const int m = q0 + (lane >> 4) * 8 + r;
#pragma unroll
    for (int j = 0; j < 4; ++j) {
      const int n = 16 * j + (lane & 15);
      ctx[(size_t)(b * S + m) * D + h * Dh + n] = (bf16)(o[j][r] * inv);
    }
  }
}

// ---------------------------------------------------------------------------
extern "C" void kernel_launch(void* const* d_in, const int* in_sizes, int n_in,
                              void* d_out, int out_size, void* d_ws, size_t ws_size,
                              hipStream_t stream) {
  (void)in_sizes; (void)n_in; (void)out_size; (void)ws_size;
  const float* x  = (const float*)d_in[0];
  const float* wq = (const float*)d_in[1];
  const float* bq = (const float*)d_in[2];
  const float* wk = (const float*)d_in[3];
  const float* bk = (const float*)d_in[4];
  const float* wv = (const float*)d_in[5];
  const float* bv = (const float*)d_in[6];
  const float* wo = (const float*)d_in[7];
  const float* bo = (const float*)d_in[8];

  constexpr int M = 4 * 2048;
  constexpr int Dm = 1024;

  bf16* p    = (bf16*)d_ws;
  bf16* xb   = p; p += (size_t)M * Dm;
  bf16* wqb  = p; p += (size_t)Dm * Dm;
  bf16* wkb  = p; p += (size_t)Dm * Dm;
  bf16* wvb  = p; p += (size_t)Dm * Dm;
  bf16* wob  = p; p += (size_t)Dm * Dm;
  bf16* qb   = p; p += (size_t)M * Dm;
  bf16* kb   = p; p += (size_t)M * Dm;
  bf16* vtb  = p; p += (size_t)M * Dm;   // [B,H,Dh,S]
  bf16* ctxb = p; p += (size_t)M * Dm;   // total ~92.3 MB

  cvt_f32_bf16<<<(M * Dm / 8 + 255) / 256, 256, 0, stream>>>(x, xb, M * Dm);
  cvt_f32_bf16<<<(Dm * Dm / 8 + 255) / 256, 256, 0, stream>>>(wq, wqb, Dm * Dm);
  cvt_f32_bf16<<<(Dm * Dm / 8 + 255) / 256, 256, 0, stream>>>(wk, wkb, Dm * Dm);
  cvt_f32_bf16<<<(Dm * Dm / 8 + 255) / 256, 256, 0, stream>>>(wv, wvb, Dm * Dm);
  cvt_f32_bf16<<<(Dm * Dm / 8 + 255) / 256, 256, 0, stream>>>(wo, wob, Dm * Dm);

  dim3 ggrid(Dm / 128, M / 128);
  gemm_bf16_wmma<0><<<ggrid, 256, 0, stream>>>(xb, wqb, bq, qb);
  gemm_bf16_wmma<0><<<ggrid, 256, 0, stream>>>(xb, wkb, bk, kb);
  gemm_bf16_wmma<1><<<ggrid, 256, 0, stream>>>(xb, wvb, bv, vtb);

  flash_attn_wmma<<<dim3(64, 16), 256, 0, stream>>>(qb, kb, vtb, ctxb);

  gemm_bf16_wmma<2><<<ggrid, 256, 0, stream>>>(ctxb, wob, bo, (float*)d_out);
}
